// MultiHeadSelfAttention_13426067768122
// MI455X (gfx1250) — matrix operand order repaired and checked by reading, not compiled, not run on hardware
//
#include <hip/hip_runtime.h>

// ---------------------------------------------------------------------------
// MultiHeadSelfAttention with RoPE for MI455X (gfx1250), wave32 + WMMA.
// Pipeline: f32->f16 convert, QKV WMMA GEMMs (software-pipelined), RoPE,
// V transpose, flash-attention with WMMA QK^T / PV, output WMMA GEMM.
// ---------------------------------------------------------------------------

typedef __attribute__((ext_vector_type(16))) _Float16 v16h;
typedef __attribute__((ext_vector_type(8)))  _Float16 v8h;
typedef __attribute__((ext_vector_type(8)))  float    v8f;

#define D_MODEL   1024
#define NUM_HEADS 16
#define HEAD_DIM  64
#define BATCH     2
#define SEQ       2048
#define MTOT      (BATCH * SEQ)          // 4096 token rows

static __device__ __forceinline__ v8f wmma_f16(v16h a, v16h b, v8f c) {
  // D = A(16x32 f16) * B(32x16 f16) + C(16x16 f32)
  return __builtin_amdgcn_wmma_f32_16x16x32_f16(
      /*neg_a=*/false, a, /*neg_b=*/false, b,
      /*c_mod=*/(short)0, c, /*reuse_a=*/false, /*reuse_b=*/false);
}

// A fragment 16x32 (MxK), row-major source, ld in f16 elements.
// lane L: row M=L&15, K = {koff..koff+7, koff+16..koff+23}, koff=(L>=16)*8
static __device__ __forceinline__ v16h load_frag_a(const _Float16* base, int ld, int lane) {
  int row  = lane & 15;
  int koff = (lane >> 4) << 3;
  const _Float16* p = base + (size_t)row * ld + koff;
  union { v16h v; v8h h[2]; } u;
  u.h[0] = *(const v8h*)(p);
  u.h[1] = *(const v8h*)(p + 16);
  return u.v;
}

// B fragment 32x16 (KxN) given B^T rows (row-major, K contiguous).
// lane L: col N = L&15, K = {koff..koff+7, koff+16..koff+23}, koff=(L>=16)*8
static __device__ __forceinline__ v16h load_frag_b(const _Float16* base, int ld, int lane) {
  int col  = lane & 15;
  int koff = (lane >> 4) << 3;
  const _Float16* p = base + (size_t)col * ld + koff;
  union { v16h v; v8h h[2]; } u;
  u.h[0] = *(const v8h*)(p);
  u.h[1] = *(const v8h*)(p + 16);
  return u.v;
}

// ---------------------------------------------------------------------------
// Vectorized f32 -> f16 convert: 8 elements per thread (n must be mult of 8).
__global__ void cvt_f32_to_f16(const float* __restrict__ src,
                               _Float16* __restrict__ dst, int n) {
  int i = (blockIdx.x * blockDim.x + threadIdx.x) * 8;
  if (i >= n) return;
  float4 f0 = *(const float4*)(src + i);
  float4 f1 = *(const float4*)(src + i + 4);
  v8h o;
  o[0] = (_Float16)f0.x; o[1] = (_Float16)f0.y;
  o[2] = (_Float16)f0.z; o[3] = (_Float16)f0.w;
  o[4] = (_Float16)f1.x; o[5] = (_Float16)f1.y;
  o[6] = (_Float16)f1.z; o[7] = (_Float16)f1.w;
  *(v8h*)(dst + i) = o;
}

// ---------------------------------------------------------------------------
// C[M,N] = A[M,K] * Bt[N,K]^T   (A,Bt f16 row-major; C f16 or f32)
// block = 128 threads (4 waves); wave -> 32 rows x 64 cols; WG -> 128x64 tile.
// Software-pipelined: next k-slice fragments load while current WMMAs run.
template <bool F32OUT>
__global__ void __launch_bounds__(128)
gemm_wmma(const _Float16* __restrict__ A, const _Float16* __restrict__ Bt,
          void* __restrict__ Cout, int M, int N, int K) {
  int lane = threadIdx.x & 31;
  int wave = threadIdx.x >> 5;
  int m0 = blockIdx.y * 128 + wave * 32;
  int n0 = blockIdx.x * 64;

  const _Float16* Ap0 = A + (size_t)m0 * K;
  const _Float16* Ap1 = A + (size_t)(m0 + 16) * K;
  const _Float16* Bp0 = Bt + (size_t)(n0 +  0) * K;
  const _Float16* Bp1 = Bt + (size_t)(n0 + 16) * K;
  const _Float16* Bp2 = Bt + (size_t)(n0 + 32) * K;
  const _Float16* Bp3 = Bt + (size_t)(n0 + 48) * K;

  v8f acc[2][4] = {};

  // prologue: k = 0 fragments
  v16h a0 = load_frag_a(Ap0, K, lane);
  v16h a1 = load_frag_a(Ap1, K, lane);
  v16h b0 = load_frag_b(Bp0, K, lane);
  v16h b1 = load_frag_b(Bp1, K, lane);
  v16h b2 = load_frag_b(Bp2, K, lane);
  v16h b3 = load_frag_b(Bp3, K, lane);

  for (int k0 = 32; k0 < K; k0 += 32) {
    // issue next-slice loads first (independent of current WMMAs)
    v16h an0 = load_frag_a(Ap0 + k0, K, lane);
    v16h an1 = load_frag_a(Ap1 + k0, K, lane);
    v16h bn0 = load_frag_b(Bp0 + k0, K, lane);
    v16h bn1 = load_frag_b(Bp1 + k0, K, lane);
    v16h bn2 = load_frag_b(Bp2 + k0, K, lane);
    v16h bn3 = load_frag_b(Bp3 + k0, K, lane);

    acc[0][0] = wmma_f16(a0, b0, acc[0][0]);
    acc[0][1] = wmma_f16(a0, b1, acc[0][1]);
    acc[0][2] = wmma_f16(a0, b2, acc[0][2]);
    acc[0][3] = wmma_f16(a0, b3, acc[0][3]);
    acc[1][0] = wmma_f16(a1, b0, acc[1][0]);
    acc[1][1] = wmma_f16(a1, b1, acc[1][1]);
    acc[1][2] = wmma_f16(a1, b2, acc[1][2]);
    acc[1][3] = wmma_f16(a1, b3, acc[1][3]);

    a0 = an0; a1 = an1;
    b0 = bn0; b1 = bn1; b2 = bn2; b3 = bn3;
  }
  // epilogue: last slice
  acc[0][0] = wmma_f16(a0, b0, acc[0][0]);
  acc[0][1] = wmma_f16(a0, b1, acc[0][1]);
  acc[0][2] = wmma_f16(a0, b2, acc[0][2]);
  acc[0][3] = wmma_f16(a0, b3, acc[0][3]);
  acc[1][0] = wmma_f16(a1, b0, acc[1][0]);
  acc[1][1] = wmma_f16(a1, b1, acc[1][1]);
  acc[1][2] = wmma_f16(a1, b2, acc[1][2]);
  acc[1][3] = wmma_f16(a1, b3, acc[1][3]);

  // C 16x16 layout: VGPR e -> (M=e, N=lane) lanes 0-15; (M=e+8, N=lane-16)
  int col = lane & 15;
  int rb  = (lane >> 4) << 3;
#pragma unroll
  for (int s = 0; s < 2; ++s) {
#pragma unroll
    for (int t = 0; t < 4; ++t) {
#pragma unroll
      for (int e = 0; e < 8; ++e) {
        size_t off = (size_t)(m0 + s * 16 + rb + e) * N + (n0 + t * 16 + col);
        if (F32OUT) ((float*)Cout)[off] = acc[s][t][e];
        else        ((_Float16*)Cout)[off] = (_Float16)acc[s][t][e];
      }
    }
  }
}

// ---------------------------------------------------------------------------
// RoPE on Q and K in place. One thread per (bt, h, pair i).
__global__ void rope_kernel(_Float16* __restrict__ Q, _Float16* __restrict__ K,
                            const int* __restrict__ pos) {
  int idx = blockIdx.x * blockDim.x + threadIdx.x;
  if (idx >= MTOT * NUM_HEADS * (HEAD_DIM / 2)) return;
  int i    = idx % (HEAD_DIM / 2);
  int rest = idx / (HEAD_DIM / 2);
  int h    = rest % NUM_HEADS;
  int bt   = rest / NUM_HEADS;

  float p    = (float)pos[bt];
  // inv_freq = theta^(-2i/hd), ln(10000) = 9.210340371976184
  float freq = __expf(-9.210340371976184f * (2.0f * (float)i) / (float)HEAD_DIM);
  float ang  = p * freq;
  float s, c;
  __sincosf(ang, &s, &c);

  size_t off = (size_t)bt * D_MODEL + h * HEAD_DIM + 2 * i;
  float q0 = (float)Q[off], q1 = (float)Q[off + 1];
  Q[off]     = (_Float16)(c * q0 - s * q1);
  Q[off + 1] = (_Float16)(c * q1 + s * q0);
  float k0 = (float)K[off], k1 = (float)K[off + 1];
  K[off]     = (_Float16)(c * k0 - s * k1);
  K[off + 1] = (_Float16)(c * k1 + s * k0);
}

// ---------------------------------------------------------------------------
// VT[b,h,d,t] = V[(b*SEQ+t)*D + h*HD + d]
__global__ void transpose_v(const _Float16* __restrict__ V,
                            _Float16* __restrict__ VT) {
  int idx = blockIdx.x * blockDim.x + threadIdx.x;
  if (idx >= BATCH * NUM_HEADS * HEAD_DIM * SEQ) return;
  int t  = idx % SEQ;
  int r  = idx / SEQ;
  int d  = r % HEAD_DIM;
  int r2 = r / HEAD_DIM;
  int h  = r2 % NUM_HEADS;
  int b  = r2 / NUM_HEADS;
  VT[idx] = V[(size_t)(b * SEQ + t) * D_MODEL + h * HEAD_DIM + d];
}

// ---------------------------------------------------------------------------
// Causal flash attention. block = 128 (4 waves); wave -> 16 query rows.
// grid.x = B*H*(SEQ/64). O written as [bt, h*64+d] f16.
__global__ void __launch_bounds__(128)
attn_kernel(const _Float16* __restrict__ Q, const _Float16* __restrict__ K,
            const _Float16* __restrict__ VT, _Float16* __restrict__ O) {
  __shared__ float    sS[4][16 * 32];   // raw scores per wave
  __shared__ _Float16 sP[4][16 * 32];   // exp()'d probs per wave
  __shared__ float    sB[4][16];        // per-row broadcast (alpha / 1/l)

  int lane = threadIdx.x & 31;
  int wave = threadIdx.x >> 5;
  int qt = blockIdx.x % (SEQ / 64);
  int bh = blockIdx.x / (SEQ / 64);
  int b = bh / NUM_HEADS, h = bh % NUM_HEADS;

  int qb0 = qt * 64 + wave * 16;                         // query strip base
  size_t rowbase = (size_t)(b * SEQ) * D_MODEL + h * HEAD_DIM;
  const _Float16* vtb = VT + (size_t)(b * NUM_HEADS + h) * HEAD_DIM * SEQ;

  v16h aq0 = load_frag_a(Q + rowbase + (size_t)qb0 * D_MODEL + 0,  D_MODEL, lane);
  v16h aq1 = load_frag_a(Q + rowbase + (size_t)qb0 * D_MODEL + 32, D_MODEL, lane);

  v8f o[4] = {};
  float m_r = -1e30f, l_r = 0.0f;
  const float scale = 0.125f;                            // 1/sqrt(64)
  int qmax = qb0 + 15;

  float*    Srow = &sS[wave][0];
  _Float16* Prow = &sP[wave][0];
  int col = lane & 15;
  int rb  = (lane >> 4) << 3;

  for (int kb = 0; kb <= qmax; kb += 32) {
    // ---- all K and V fragments up front: 16 b128 loads in flight while the
    //      score WMMAs and the serial softmax scan execute.
    v16h bk0 = load_frag_b(K + rowbase + (size_t)(kb + 0)  * D_MODEL + 0,  D_MODEL, lane);
    v16h bk1 = load_frag_b(K + rowbase + (size_t)(kb + 0)  * D_MODEL + 32, D_MODEL, lane);
    v16h bk2 = load_frag_b(K + rowbase + (size_t)(kb + 16) * D_MODEL + 0,  D_MODEL, lane);
    v16h bk3 = load_frag_b(K + rowbase + (size_t)(kb + 16) * D_MODEL + 32, D_MODEL, lane);
    v16h bv0 = load_frag_b(vtb + (size_t) 0 * SEQ + kb, SEQ, lane);
    v16h bv1 = load_frag_b(vtb + (size_t)16 * SEQ + kb, SEQ, lane);
    v16h bv2 = load_frag_b(vtb + (size_t)32 * SEQ + kb, SEQ, lane);
    v16h bv3 = load_frag_b(vtb + (size_t)48 * SEQ + kb, SEQ, lane);

    // prefetch next key block (K rows + VT rows) into near caches (loc=3 ->
    // WGP-scope prefetch, pulls into all cache levels)
    if (kb + 32 <= qmax) {
      __builtin_prefetch(K + rowbase + (size_t)(kb + 32 + lane) * D_MODEL, 0, 3);
      __builtin_prefetch(vtb + (size_t)lane * SEQ + (kb + 32), 0, 3);
      __builtin_prefetch(vtb + (size_t)(32 + lane) * SEQ + (kb + 32), 0, 3);
    }

    v8f s0 = {}, s1 = {};
    s0 = wmma_f16(aq0, bk0, s0);
    s0 = wmma_f16(aq1, bk1, s0);
    s1 = wmma_f16(aq0, bk2, s1);
    s1 = wmma_f16(aq1, bk3, s1);

    // ---- spill scores to LDS (per-wave region; intra-wave DS is in-order)
#pragma unroll
    for (int e = 0; e < 8; ++e) {
      Srow[(rb + e) * 32 + col]      = s0[e];
      Srow[(rb + e) * 32 + col + 16] = s1[e];
    }
    __builtin_amdgcn_wave_barrier();

    // ---- online softmax scan: lane r (<16) owns query row qb0+r
    if (lane < 16) {
      int qrow = qb0 + lane;
      int cmax = qrow - kb; if (cmax > 31) cmax = 31;    // causal clamp
      float mnew = m_r;
      for (int c2 = 0; c2 <= cmax; ++c2) {
        float v = Srow[lane * 32 + c2] * scale;
        mnew = fmaxf(mnew, v);
      }
      float alpha = __expf(m_r - mnew);
      float lsum  = l_r * alpha;
      for (int c2 = 0; c2 < 32; ++c2) {
        float v = Srow[lane * 32 + c2] * scale;
        float p = (c2 <= cmax) ? __expf(v - mnew) : 0.0f;
        Prow[lane * 32 + c2] = (_Float16)p;
        lsum += p;
      }
      m_r = mnew; l_r = lsum;
      sB[wave][lane] = alpha;
    }
    __builtin_amdgcn_wave_barrier();

    // ---- rescale accumulators by per-row alpha (LDS broadcast)
#pragma unroll
    for (int e = 0; e < 8; ++e) {
      float al = sB[wave][rb + e];
      o[0][e] *= al; o[1][e] *= al; o[2][e] *= al; o[3][e] *= al;
    }

    // ---- P fragment from LDS (16x32 f16, A layout)
    v16h ap;
    {
      int prow = lane & 15;
      int koff = (lane >> 4) << 3;
      union { v16h v; v8h hx[2]; } u;
      u.hx[0] = *(const v8h*)&Prow[prow * 32 + koff];
      u.hx[1] = *(const v8h*)&Prow[prow * 32 + koff + 16];
      ap = u.v;
    }

    // ---- O += P(16x32) * V(32x64)
    o[0] = wmma_f16(ap, bv0, o[0]);
    o[1] = wmma_f16(ap, bv1, o[1]);
    o[2] = wmma_f16(ap, bv2, o[2]);
    o[3] = wmma_f16(ap, bv3, o[3]);
  }

  // ---- normalize by 1/l and store O f16 as [bt, h*64 + d]
  if (lane < 16) sB[wave][lane] = 1.0f / l_r;
  __builtin_amdgcn_wave_barrier();
#pragma unroll
  for (int e = 0; e < 8; ++e) {
    float inv = sB[wave][rb + e];
    size_t out0 = rowbase + (size_t)(qb0 + rb + e) * D_MODEL;
#pragma unroll
    for (int t2 = 0; t2 < 4; ++t2)
      O[out0 + t2 * 16 + col] = (_Float16)(o[t2][e] * inv);
  }
}

// ---------------------------------------------------------------------------
extern "C" void kernel_launch(void* const* d_in, const int* in_sizes, int n_in,
                              void* d_out, int out_size, void* d_ws, size_t ws_size,
                              hipStream_t stream) {
  (void)in_sizes; (void)n_in; (void)out_size; (void)ws_size;

  const float* X   = (const float*)d_in[0];
  const int*   pos = (const int*)  d_in[1];
  const float* Wq  = (const float*)d_in[2];
  const float* Wk  = (const float*)d_in[3];
  const float* Wv  = (const float*)d_in[4];
  const float* Wo  = (const float*)d_in[5];
  float* out = (float*)d_out;

  // Workspace layout (f16 buffers), total 56 MB.
  const size_t MB = 1u << 20;
  char* ws = (char*)d_ws;
  _Float16* Xh  = (_Float16*)(ws + 0 * MB);    // 4096x1024 (8MB)
  _Float16* Wqh = (_Float16*)(ws + 8 * MB);    // 1024x1024 (2MB)
  _Float16* Wkh = (_Float16*)(ws + 10 * MB);
  _Float16* Wvh = (_Float16*)(ws + 12 * MB);
  _Float16* Woh = (_Float16*)(ws + 14 * MB);
  _Float16* Qh  = (_Float16*)(ws + 16 * MB);   // 8MB
  _Float16* Kh  = (_Float16*)(ws + 24 * MB);
  _Float16* Vh  = (_Float16*)(ws + 32 * MB);
  _Float16* VTh = (_Float16*)(ws + 40 * MB);
  _Float16* Oh  = (_Float16*)(ws + 48 * MB);

  // 1) convert f32 -> f16 (8 elems/thread)
  {
    int nX = MTOT * D_MODEL;        // 4194304
    int nW = D_MODEL * D_MODEL;     // 1048576
    cvt_f32_to_f16<<<nX / 8 / 256, 256, 0, stream>>>(X,  Xh,  nX);
    cvt_f32_to_f16<<<nW / 8 / 256, 256, 0, stream>>>(Wq, Wqh, nW);
    cvt_f32_to_f16<<<nW / 8 / 256, 256, 0, stream>>>(Wk, Wkh, nW);
    cvt_f32_to_f16<<<nW / 8 / 256, 256, 0, stream>>>(Wv, Wvh, nW);
    cvt_f32_to_f16<<<nW / 8 / 256, 256, 0, stream>>>(Wo, Woh, nW);
  }

  // 2) QKV projections: [4096,1024] = Xh * W^T
  dim3 ggrid(D_MODEL / 64, MTOT / 128);   // (16, 32)
  gemm_wmma<false><<<ggrid, 128, 0, stream>>>(Xh, Wqh, Qh, MTOT, D_MODEL, D_MODEL);
  gemm_wmma<false><<<ggrid, 128, 0, stream>>>(Xh, Wkh, Kh, MTOT, D_MODEL, D_MODEL);
  gemm_wmma<false><<<ggrid, 128, 0, stream>>>(Xh, Wvh, Vh, MTOT, D_MODEL, D_MODEL);

  // 3) RoPE on Q, K
  {
    int n = MTOT * NUM_HEADS * (HEAD_DIM / 2);   // 2097152
    rope_kernel<<<n / 256, 256, 0, stream>>>(Qh, Kh, pos);
  }

  // 4) V -> VT[b,h,d,t]
  {
    int n = BATCH * NUM_HEADS * HEAD_DIM * SEQ;  // 4194304
    transpose_v<<<n / 256, 256, 0, stream>>>(Vh, VTh);
  }

  // 5) causal flash attention
  attn_kernel<<<BATCH * NUM_HEADS * (SEQ / 64), 128, 0, stream>>>(Qh, Kh, VTh, Oh);

  // 6) output projection (f32 store into d_out)
  gemm_wmma<true><<<ggrid, 128, 0, stream>>>(Oh, Woh, out, MTOT, D_MODEL, D_MODEL);
}
